// ArcticDecoderLayer_43447889166549
// MI455X (gfx1250) — matrix operand order repaired and checked by reading, not compiled, not run on hardware
//
#include <hip/hip_runtime.h>
#include <hip/hip_bf16.h>

// ---------------- problem constants ----------------
#define T_TOK 2048
#define H_DIM 1024
#define S_LEN 1024
#define NH_Q  16
#define NKV_H 4
#define HD_D  64
#define F_DIM 2816
#define E_EXP 8
#define WIN_SZ 256
#define EPS_F 1e-5f

typedef __bf16 bhalf;
typedef __attribute__((ext_vector_type(16))) __bf16 v16bf;
typedef __attribute__((ext_vector_type(2)))  __bf16 v2bf;
typedef __attribute__((ext_vector_type(8)))  float  v8f;

__device__ __forceinline__ bhalf f2bf(float f) { return (bhalf)f; }
__device__ __forceinline__ float bf2f(bhalf b) { return (float)b; }
__device__ __forceinline__ unsigned pack2f(float a, float b) {
  v2bf p; p[0] = (bhalf)a; p[1] = (bhalf)b;
  return __builtin_bit_cast(unsigned, p);
}

__device__ __forceinline__ v8f wmma_bf16(v16bf a, v16bf b, v8f c) {
  // D(16x16,f32) = A(16x32,bf16) x B(32x16,bf16) + C
  return __builtin_amdgcn_wmma_f32_16x16x32_bf16(false, a, false, b, (short)0, c, false, false);
}

// ---- CDNA5 async global->LDS copy (ASYNCcnt-tracked) ----
__device__ __forceinline__ void async_ld_b128(unsigned lds_off, const void* g) {
  asm volatile("global_load_async_to_lds_b128 %0, %1, off"
               :: "v"(lds_off), "v"((unsigned long long)(size_t)g) : "memory");
}
__device__ __forceinline__ void wait_async0() {
  asm volatile("s_wait_asynccnt 0x0" ::: "memory");
}
__device__ __forceinline__ void wait_ds0() {
  asm volatile("s_wait_dscnt 0x0" ::: "memory");
}

// A fragment: 16x32 (MxK) from row-major src[m*ld + k].
// Element i -> k = 16*half + i: contiguous 16 halfwords -> 2x ds_load_b128 when ld%8==0.
__device__ __forceinline__ v16bf load_frag_a(const bhalf* src, int ld) {
  int lane = threadIdx.x & 31;
  int half = lane >> 4, m = lane & 15;
  v16bf r;
#pragma unroll
  for (int i = 0; i < 16; ++i) {
    int k = (half << 4) + i;
    r[i] = src[m * ld + k];
  }
  return r;
}
// B fragment: 32x16 (KxN) from N-major storage src[n*ld + k].
// Element i -> k = 16*half + i: contiguous 16 halfwords -> 2x ds_load_b128.
__device__ __forceinline__ v16bf load_frag_b_nm(const bhalf* src, int ld) {
  int lane = threadIdx.x & 31;
  int half = lane >> 4, n = lane & 15;
  v16bf r;
#pragma unroll
  for (int i = 0; i < 16; ++i) r[i] = src[n * ld + (half << 4) + i];
  return r;
}

// ---------------- RMSNorm: f32 in -> bf16 out ----------------
__global__ void __launch_bounds__(256)
rmsnorm_kernel(const float* __restrict__ x, const float* __restrict__ w,
               bhalf* __restrict__ out) {
  int t = blockIdx.x;
  const float* xr = x + (size_t)t * H_DIM;
  __shared__ float red[8];
  float ss = 0.f;
  for (int i = threadIdx.x; i < H_DIM; i += 256) { float v = xr[i]; ss += v * v; }
  for (int o = 16; o > 0; o >>= 1) ss += __shfl_down(ss, o, 32);
  if ((threadIdx.x & 31) == 0) red[threadIdx.x >> 5] = ss;
  __syncthreads();
  if (threadIdx.x == 0) {
    float s = 0.f;
#pragma unroll
    for (int i = 0; i < 8; ++i) s += red[i];
    red[0] = rsqrtf(s / (float)H_DIM + EPS_F);
  }
  __syncthreads();
  float r = red[0];
  for (int i = threadIdx.x; i < H_DIM; i += 256)
    out[(size_t)t * H_DIM + i] = f2bf(xr[i] * r * w[i]);
}

// ---------------- tiling ----------------
#define BM 128
#define BN 64
#define BK 32
#define LDA 40   // A tile row stride (elems), mult of 8
#define LDBT 40  // N-major B tile row stride (elems, k-dim), mult of 8

// async-stage one thread's 16 bf16 of the A tile (two b128 transfers)
__device__ __forceinline__ void stage_a_async(const bhalf* g, bhalf* dst) {
  unsigned lo = (unsigned)(size_t)dst;
  async_ld_b128(lo, g);
  async_ld_b128(lo + 16, g + 8);
}

// stage a 32(k) x 64(n) f32 weight tile into N-major bf16 LDS tile
__device__ __forceinline__ void stage_b_nm(const float* __restrict__ Bm, int ldb,
                                           int k0, int bn, bhalf* sBt, int tid) {
  int bn2 = (tid & 31) * 2;   // n offset (pairs)
  int bk4 = (tid >> 5) * 4;   // k offset (quads)
  const float* base = Bm + (size_t)(k0 + bk4) * ldb + bn + bn2;
  float2 r0 = *reinterpret_cast<const float2*>(base);
  float2 r1 = *reinterpret_cast<const float2*>(base + ldb);
  float2 r2 = *reinterpret_cast<const float2*>(base + 2 * (size_t)ldb);
  float2 r3 = *reinterpret_cast<const float2*>(base + 3 * (size_t)ldb);
  uint2 c0, c1;
  c0.x = pack2f(r0.x, r1.x);
  c0.y = pack2f(r2.x, r3.x);
  c1.x = pack2f(r0.y, r1.y);
  c1.y = pack2f(r2.y, r3.y);
  *reinterpret_cast<uint2*>(&sBt[(bn2 + 0) * LDBT + bk4]) = c0;
  *reinterpret_cast<uint2*>(&sBt[(bn2 + 1) * LDBT + bk4]) = c1;
}

// ---------------- generic WMMA GEMM: C = A(bf16) x B(f32 weights) [+ addC] ----------------
__global__ void __launch_bounds__(256)
gemm_kernel(const bhalf* __restrict__ A, int lda,
            const float* __restrict__ Bm, int ldb,
            const float* __restrict__ addC,
            float* __restrict__ C, int ldc,
            int N, int K) {
  __shared__ bhalf sA[2][BM * LDA];
  __shared__ bhalf sBt[2][BN * LDBT];
  int bm = blockIdx.y * BM, bn = blockIdx.x * BN;
  int tid = threadIdx.x, lane = tid & 31, wave = tid >> 5;
  int wm = (wave >> 1) * 32, wn = (wave & 1) * 32;
  v8f acc[2][2];
#pragma unroll
  for (int i = 0; i < 2; ++i)
#pragma unroll
    for (int j = 0; j < 2; ++j) acc[i][j] = (v8f){0,0,0,0,0,0,0,0};
  int arow = tid >> 1, acol = (tid & 1) * 16;
  int bn2 = (tid & 31) * 2, bk4 = (tid >> 5) * 4;
  const bhalf* agp = A + (size_t)(bm + arow) * lda + acol;
  bhalf* adst = &sA[0][0] + arow * LDA + acol;
  // prologue: stage tile 0 into buffer 0
  stage_a_async(agp, adst);
  stage_b_nm(Bm, ldb, 0, bn, sBt[0], tid);
  int buf = 0;
  for (int k0 = 0; k0 < K; k0 += BK) {
    wait_async0();
    __syncthreads();
    int nk = k0 + BK;
    if (nk < K) {
      stage_a_async(agp + nk, &sA[buf ^ 1][0] + arow * LDA + acol);
      stage_b_nm(Bm, ldb, nk, bn, sBt[buf ^ 1], tid);
    }
    if (k0 + 2 * BK < K)
      __builtin_prefetch(Bm + (size_t)(k0 + 2 * BK + bk4) * ldb + bn + bn2, 0, 1);
    const bhalf* cA = sA[buf];
    const bhalf* cB = sBt[buf];
    v16bf a0 = load_frag_a(&cA[(wm +  0) * LDA], LDA);
    v16bf a1 = load_frag_a(&cA[(wm + 16) * LDA], LDA);
    v16bf b0 = load_frag_b_nm(&cB[(wn +  0) * LDBT], LDBT);
    v16bf b1 = load_frag_b_nm(&cB[(wn + 16) * LDBT], LDBT);
    acc[0][0] = wmma_bf16(a0, b0, acc[0][0]);
    acc[0][1] = wmma_bf16(a0, b1, acc[0][1]);
    acc[1][0] = wmma_bf16(a1, b0, acc[1][0]);
    acc[1][1] = wmma_bf16(a1, b1, acc[1][1]);
    buf ^= 1;
  }
  int half = lane >> 4, ncol = lane & 15;
#pragma unroll
  for (int im = 0; im < 2; ++im)
#pragma unroll
    for (int in = 0; in < 2; ++in)
#pragma unroll
      for (int r = 0; r < 8; ++r) {
        int m = bm + wm + im * 16 + r + 8 * half;
        int n = bn + wn + in * 16 + ncol;
        size_t idx = (size_t)m * ldc + n;
        float v = acc[im][in][r];
        if (addC) v += addC[idx];
        C[idx] = v;
      }
}

// -------- fused dual GEMM + SwiGLU (dense MLP and gathered MoE GEMM1) --------
__global__ void __launch_bounds__(256)
dual_gemm_silu_kernel(const bhalf* __restrict__ A, int lda,
                      const int* __restrict__ perm, const int* __restrict__ eoff,
                      const int* __restrict__ ecnt,
                      const float* __restrict__ B1g, const float* __restrict__ B3g,
                      unsigned long long bstride,
                      bhalf* __restrict__ Out, int ldo,
                      int Mstatic, int N, int K) {
  __shared__ bhalf sA[2][BM * LDA];
  __shared__ bhalf sB1t[2][BN * LDBT];
  __shared__ bhalf sB3t[2][BN * LDBT];
  int e = blockIdx.z;
  int M = ecnt ? ecnt[e] : Mstatic;
  int bm = blockIdx.y * BM, bn = blockIdx.x * BN;
  if (bm >= M) return;
  int outbase = eoff ? eoff[e] : 0;
  const float* B1 = B1g + (size_t)e * bstride;
  const float* B3 = B3g + (size_t)e * bstride;
  int tid = threadIdx.x, lane = tid & 31, wave = tid >> 5;
  int wm = (wave >> 1) * 32, wn = (wave & 1) * 32;
  v8f acc1[2][2], acc3[2][2];
#pragma unroll
  for (int i = 0; i < 2; ++i)
#pragma unroll
    for (int j = 0; j < 2; ++j) { acc1[i][j] = (v8f){0,0,0,0,0,0,0,0}; acc3[i][j] = (v8f){0,0,0,0,0,0,0,0}; }
  int arow = tid >> 1, acol = (tid & 1) * 16;
  int bn2 = (tid & 31) * 2, bk4 = (tid >> 5) * 4;
  int grow = bm + arow;
  int srow = (grow < M) ? (perm ? perm[outbase + grow] : grow) : 0;
  const bhalf* agp = A + (size_t)srow * lda + acol;
  stage_a_async(agp, &sA[0][0] + arow * LDA + acol);
  stage_b_nm(B1, N, 0, bn, sB1t[0], tid);
  stage_b_nm(B3, N, 0, bn, sB3t[0], tid);
  int buf = 0;
  for (int k0 = 0; k0 < K; k0 += BK) {
    wait_async0();
    __syncthreads();
    int nk = k0 + BK;
    if (nk < K) {
      stage_a_async(agp + nk, &sA[buf ^ 1][0] + arow * LDA + acol);
      stage_b_nm(B1, N, nk, bn, sB1t[buf ^ 1], tid);
      stage_b_nm(B3, N, nk, bn, sB3t[buf ^ 1], tid);
    }
    if (k0 + 2 * BK < K) {
      __builtin_prefetch(B1 + (size_t)(k0 + 2 * BK + bk4) * N + bn + bn2, 0, 1);
      __builtin_prefetch(B3 + (size_t)(k0 + 2 * BK + bk4) * N + bn + bn2, 0, 1);
    }
    const bhalf* cA = sA[buf];
    v16bf a0 = load_frag_a(&cA[(wm +  0) * LDA], LDA);
    v16bf a1 = load_frag_a(&cA[(wm + 16) * LDA], LDA);
    v16bf b10 = load_frag_b_nm(&sB1t[buf][(wn +  0) * LDBT], LDBT);
    v16bf b11 = load_frag_b_nm(&sB1t[buf][(wn + 16) * LDBT], LDBT);
    v16bf b30 = load_frag_b_nm(&sB3t[buf][(wn +  0) * LDBT], LDBT);
    v16bf b31 = load_frag_b_nm(&sB3t[buf][(wn + 16) * LDBT], LDBT);
    acc1[0][0] = wmma_bf16(a0, b10, acc1[0][0]);
    acc1[0][1] = wmma_bf16(a0, b11, acc1[0][1]);
    acc1[1][0] = wmma_bf16(a1, b10, acc1[1][0]);
    acc1[1][1] = wmma_bf16(a1, b11, acc1[1][1]);
    acc3[0][0] = wmma_bf16(a0, b30, acc3[0][0]);
    acc3[0][1] = wmma_bf16(a0, b31, acc3[0][1]);
    acc3[1][0] = wmma_bf16(a1, b30, acc3[1][0]);
    acc3[1][1] = wmma_bf16(a1, b31, acc3[1][1]);
    buf ^= 1;
  }
  int half = lane >> 4, ncol = lane & 15;
#pragma unroll
  for (int im = 0; im < 2; ++im)
#pragma unroll
    for (int in = 0; in < 2; ++in)
#pragma unroll
      for (int r = 0; r < 8; ++r) {
        int m = bm + wm + im * 16 + r + 8 * half;
        if (m >= M) continue;
        int n = bn + wn + in * 16 + ncol;
        float x1 = acc1[im][in][r], x3 = acc3[im][in][r];
        float hval = (x1 / (1.f + __expf(-x1))) * x3;
        Out[(size_t)(outbase + m) * ldo + n] = f2bf(hval);
      }
}

// -------- MoE GEMM2: y = hmid @ w2[e], scatter out[token] += w * y --------
__global__ void __launch_bounds__(256)
moe_gemm2_kernel(const bhalf* __restrict__ A, int lda,
                 const int* __restrict__ perm, const float* __restrict__ pw,
                 const int* __restrict__ eoff, const int* __restrict__ ecnt,
                 const float* __restrict__ B2g, unsigned long long bstride,
                 float* __restrict__ Out, int ldo, int N, int K) {
  __shared__ bhalf sA[2][BM * LDA];
  __shared__ bhalf sBt[2][BN * LDBT];
  int e = blockIdx.z;
  int M = ecnt[e], base = eoff[e];
  int bm = blockIdx.y * BM, bn = blockIdx.x * BN;
  if (bm >= M) return;
  const float* B2 = B2g + (size_t)e * bstride;
  int tid = threadIdx.x, lane = tid & 31, wave = tid >> 5;
  int wm = (wave >> 1) * 32, wn = (wave & 1) * 32;
  v8f acc[2][2];
#pragma unroll
  for (int i = 0; i < 2; ++i)
#pragma unroll
    for (int j = 0; j < 2; ++j) acc[i][j] = (v8f){0,0,0,0,0,0,0,0};
  int arow = tid >> 1, acol = (tid & 1) * 16;
  int bn2 = (tid & 31) * 2, bk4 = (tid >> 5) * 4;
  int grow = bm + arow;
  int srow = base + ((grow < M) ? grow : 0);
  const bhalf* agp = A + (size_t)srow * lda + acol;
  stage_a_async(agp, &sA[0][0] + arow * LDA + acol);
  stage_b_nm(B2, N, 0, bn, sBt[0], tid);
  int buf = 0;
  for (int k0 = 0; k0 < K; k0 += BK) {
    wait_async0();
    __syncthreads();
    int nk = k0 + BK;
    if (nk < K) {
      stage_a_async(agp + nk, &sA[buf ^ 1][0] + arow * LDA + acol);
      stage_b_nm(B2, N, nk, bn, sBt[buf ^ 1], tid);
    }
    if (k0 + 2 * BK < K)
      __builtin_prefetch(B2 + (size_t)(k0 + 2 * BK + bk4) * N + bn + bn2, 0, 1);
    const bhalf* cA = sA[buf];
    const bhalf* cB = sBt[buf];
    v16bf a0 = load_frag_a(&cA[(wm +  0) * LDA], LDA);
    v16bf a1 = load_frag_a(&cA[(wm + 16) * LDA], LDA);
    v16bf b0 = load_frag_b_nm(&cB[(wn +  0) * LDBT], LDBT);
    v16bf b1 = load_frag_b_nm(&cB[(wn + 16) * LDBT], LDBT);
    acc[0][0] = wmma_bf16(a0, b0, acc[0][0]);
    acc[0][1] = wmma_bf16(a0, b1, acc[0][1]);
    acc[1][0] = wmma_bf16(a1, b0, acc[1][0]);
    acc[1][1] = wmma_bf16(a1, b1, acc[1][1]);
    buf ^= 1;
  }
  int half = lane >> 4, ncol = lane & 15;
#pragma unroll
  for (int im = 0; im < 2; ++im)
#pragma unroll
    for (int r = 0; r < 8; ++r) {
      int m = bm + wm + im * 16 + r + 8 * half;
      if (m >= M) continue;
      int tok = perm[base + m];
      float wgt = pw[base + m];
#pragma unroll
      for (int in = 0; in < 2; ++in) {
        int n = bn + wn + in * 16 + ncol;
        atomicAdd(&Out[(size_t)tok * ldo + n], wgt * acc[im][in][r]);
      }
    }
}

// ---------------- RoPE + pack to bf16 ----------------
__global__ void __launch_bounds__(256)
rope_pack_kernel(const float* __restrict__ x, bhalf* __restrict__ out, int nh, int doRope) {
  int t = blockIdx.x;
  int s = t % S_LEN;
  int E = nh * HD_D;
  const float* xr = x + (size_t)t * E;
  for (int i = threadIdx.x; i < E; i += 256) {
    int d = i & 63;
    float v = xr[i];
    float o;
    if (doRope) {
      int j = d & 31;
      float ang = (float)s * __expf(-(float)j * (9.2103403720f / 32.0f)); // theta^(-j/32)
      float c = __cosf(ang), sn = __sinf(ang);
      float other = (d < 32) ? -xr[i + 32] : xr[i - 32];
      o = v * c + other * sn;
    } else {
      o = v;
    }
    out[(size_t)t * E + i] = f2bf(o);
  }
}

// ---------------- sliding-window GQA flash attention (1 wave / 16-query tile) ----------------
#define LDV 40
__global__ void __launch_bounds__(32)
attn_kernel(const bhalf* __restrict__ qb, const bhalf* __restrict__ kb,
            const bhalf* __restrict__ vb, bhalf* __restrict__ ob) {
  __shared__ bhalf sQ[16 * 72];
  __shared__ bhalf sK[32 * 72];
  __shared__ bhalf sVt[64 * LDV];  // dim-major V^T tile: sVt[d][key]
  __shared__ bhalf sP[16 * 40];
  int qt = blockIdx.x, h = blockIdx.y, b = blockIdx.z;
  int lane = threadIdx.x;
  int q0 = qt * 16;
  int kvh = h >> 2;  // GQA: 16 q heads -> 4 kv heads
  {
    int r = lane & 15, c = (lane >> 4) * 32;
    const bhalf* src = qb + ((size_t)(b * S_LEN + q0 + r)) * (NH_Q * HD_D) + h * HD_D + c;
    unsigned lo = (unsigned)(size_t)&sQ[r * 72 + c];
    async_ld_b128(lo, src);
    async_ld_b128(lo + 16, src + 8);
    async_ld_b128(lo + 32, src + 16);
    async_ld_b128(lo + 48, src + 24);
  }
  wait_async0();
  v16bf qa0 = load_frag_a(&sQ[0], 72);
  v16bf qa1 = load_frag_a(&sQ[32], 72);
  float mrow[8], lrow[8];
  v8f accs[4];
#pragma unroll
  for (int r = 0; r < 8; ++r) { mrow[r] = -1e30f; lrow[r] = 0.f; }
#pragma unroll
  for (int c = 0; c < 4; ++c) accs[c] = (v8f){0,0,0,0,0,0,0,0};
  int half = lane >> 4, ncol = lane & 15;
  int lo = q0 - (WIN_SZ - 1); if (lo < 0) lo = 0;
  for (int kbase = lo & ~31; kbase <= q0 + 15; kbase += 32) {
    // --- K tile: async-staged row-major (fragments read contiguously in d)
    {
      int krow = kbase + lane;
      size_t roff = ((size_t)(b * S_LEN + ((krow < S_LEN) ? krow : 0))) * (NKV_H * HD_D) + kvh * HD_D;
      const bhalf* src = kb + roff;
      unsigned klo = (unsigned)(size_t)&sK[lane * 72];
      async_ld_b128(klo, src);
      async_ld_b128(klo + 16, src + 8);
      async_ld_b128(klo + 32, src + 16);
      async_ld_b128(klo + 48, src + 24);
    }
    // --- V tile: stage dim-major (key pairs packed into dwords)
    {
      int pr = lane & 15;          // key pair -> keys 2pr, 2pr+1
      int dh = (lane >> 4) * 32;   // dim half
      int ka = kbase + 2 * pr, kc = ka + 1;
      size_t ra = ((size_t)(b * S_LEN + ((ka < S_LEN) ? ka : 0))) * (NKV_H * HD_D) + kvh * HD_D + dh;
      size_t rc = ((size_t)(b * S_LEN + ((kc < S_LEN) ? kc : 0))) * (NKV_H * HD_D) + kvh * HD_D + dh;
      const uint4* pa4 = reinterpret_cast<const uint4*>(vb + ra);
      const uint4* pc4 = reinterpret_cast<const uint4*>(vb + rc);
      uint4 A0 = pa4[0], A1 = pa4[1], A2 = pa4[2], A3 = pa4[3];
      uint4 C0 = pc4[0], C1 = pc4[1], C2 = pc4[2], C3 = pc4[3];
      if (ka >= S_LEN) { A0 = A1 = A2 = A3 = (uint4){0,0,0,0}; }
      if (kc >= S_LEN) { C0 = C1 = C2 = C3 = (uint4){0,0,0,0}; }
      unsigned wa[16] = {A0.x,A0.y,A0.z,A0.w, A1.x,A1.y,A1.z,A1.w,
                         A2.x,A2.y,A2.z,A2.w, A3.x,A3.y,A3.z,A3.w};
      unsigned wc[16] = {C0.x,C0.y,C0.z,C0.w, C1.x,C1.y,C1.z,C1.w,
                         C2.x,C2.y,C2.z,C2.w, C3.x,C3.y,C3.z,C3.w};
#pragma unroll
      for (int j = 0; j < 16; ++j) {
        unsigned a = wa[j], c = wc[j];
        unsigned lo32 = (a & 0xFFFFu) | (c << 16);
        unsigned hi32 = (a >> 16) | (c & 0xFFFF0000u);
        *reinterpret_cast<unsigned*>(&sVt[(dh + 2 * j + 0) * LDV + 2 * pr]) = lo32;
        *reinterpret_cast<unsigned*>(&sVt[(dh + 2 * j + 1) * LDV + 2 * pr]) = hi32;
      }
    }
    wait_async0();
    wait_ds0();
    v8f s0 = (v8f){0,0,0,0,0,0,0,0}, s1 = (v8f){0,0,0,0,0,0,0,0};
    s0 = wmma_bf16(qa0, load_frag_b_nm(&sK[0], 72), s0);                // keys 0..15,  d 0..31
    s0 = wmma_bf16(qa1, load_frag_b_nm(&sK[32], 72), s0);               // keys 0..15,  d 32..63
    s1 = wmma_bf16(qa0, load_frag_b_nm(&sK[16 * 72], 72), s1);          // keys 16..31, d 0..31
    s1 = wmma_bf16(qa1, load_frag_b_nm(&sK[16 * 72 + 32], 72), s1);     // keys 16..31, d 32..63
    float p0[8], p1[8];
#pragma unroll
    for (int r = 0; r < 8; ++r) {
      int m = r + 8 * half;
      int qi = q0 + m;
      int k0i = kbase + ncol, k1i = kbase + 16 + ncol;
      float v0 = s0[r] * 0.125f, v1 = s1[r] * 0.125f;
      v0 = ((k0i <= qi) && (qi - k0i < WIN_SZ)) ? v0 : -1e30f;
      v1 = ((k1i <= qi) && (qi - k1i < WIN_SZ)) ? v1 : -1e30f;
      float mx = fmaxf(v0, v1);
      for (int off = 8; off; off >>= 1) mx = fmaxf(mx, __shfl_xor(mx, off, 16));
      float mnew = fmaxf(mrow[r], mx);
      float alpha = __expf(mrow[r] - mnew);
      float e0 = __expf(v0 - mnew), e1 = __expf(v1 - mnew);
      float ls = e0 + e1;
      for (int off = 8; off; off >>= 1) ls += __shfl_xor(ls, off, 16);
      lrow[r] = lrow[r] * alpha + ls;
      mrow[r] = mnew;
#pragma unroll
      for (int c = 0; c < 4; ++c) accs[c][r] *= alpha;
      p0[r] = e0; p1[r] = e1;
    }
#pragma unroll
    for (int r = 0; r < 8; ++r) {
      int m = r + 8 * half;
      sP[m * 40 + ncol] = f2bf(p0[r]);
      sP[m * 40 + 16 + ncol] = f2bf(p1[r]);
    }
    wait_ds0();
    v16bf pa = load_frag_a(&sP[0], 40);
#pragma unroll
    for (int c = 0; c < 4; ++c)
      accs[c] = wmma_bf16(pa, load_frag_b_nm(&sVt[(c * 16) * LDV], LDV), accs[c]);
  }
#pragma unroll
  for (int c = 0; c < 4; ++c)
#pragma unroll
    for (int r = 0; r < 8; ++r) {
      int m = r + 8 * half;
      float v = accs[c][r] / lrow[r];
      int tok = b * S_LEN + q0 + m;
      ob[(size_t)tok * (NH_Q * HD_D) + h * HD_D + c * 16 + ncol] = f2bf(v);
    }
}

// ---------------- MoE routing ----------------
__global__ void zero16_kernel(int* p) { if (threadIdx.x < 16) p[threadIdx.x] = 0; }

__global__ void __launch_bounds__(32)
gate_kernel(const bhalf* __restrict__ x, const float* __restrict__ gw,
            int* __restrict__ eid, float* __restrict__ ew, int* __restrict__ cnt) {
  int t = blockIdx.x, lane = threadIdx.x;
  float acc[E_EXP];
#pragma unroll
  for (int e = 0; e < E_EXP; ++e) acc[e] = 0.f;
  for (int hh = lane; hh < H_DIM; hh += 32) {
    float xv = bf2f(x[(size_t)t * H_DIM + hh]);
#pragma unroll
    for (int e = 0; e < E_EXP; ++e) acc[e] += xv * gw[hh * E_EXP + e];
  }
#pragma unroll
  for (int e = 0; e < E_EXP; ++e)
    for (int o = 16; o > 0; o >>= 1) acc[e] += __shfl_down(acc[e], o, 32);
  if (lane == 0) {
    float mx = -1e30f;
#pragma unroll
    for (int e = 0; e < E_EXP; ++e) mx = fmaxf(mx, acc[e]);
    float p[E_EXP], s = 0.f;
#pragma unroll
    for (int e = 0; e < E_EXP; ++e) { p[e] = __expf(acc[e] - mx); s += p[e]; }
    int i0 = 0;
#pragma unroll
    for (int e = 1; e < E_EXP; ++e) if (p[e] > p[i0]) i0 = e;
    int i1 = (i0 == 0) ? 1 : 0;
#pragma unroll
    for (int e = 0; e < E_EXP; ++e) if (e != i0 && p[e] > p[i1]) i1 = e;
    float w0 = p[i0] / s, w1 = p[i1] / s, tot = w0 + w1;
    eid[2 * t] = i0; eid[2 * t + 1] = i1;
    ew[2 * t] = w0 / tot; ew[2 * t + 1] = w1 / tot;
    atomicAdd(&cnt[i0], 1); atomicAdd(&cnt[i1], 1);
  }
}

__global__ void offsets_kernel(const int* cnt, int* off, int* cur) {
  if (threadIdx.x == 0) {
    int s = 0;
    for (int e = 0; e < E_EXP; ++e) { off[e] = s; s += cnt[e]; cur[e] = 0; }
  }
}

__global__ void scatter_kernel(const int* eid, const float* ew, const int* off,
                               int* cur, int* perm, float* pw) {
  int t = blockIdx.x * blockDim.x + threadIdx.x;
  if (t >= T_TOK) return;
  for (int s2 = 0; s2 < 2; ++s2) {
    int e = eid[2 * t + s2];
    int p = atomicAdd(&cur[e], 1);
    perm[off[e] + p] = t;
    pw[off[e] + p] = ew[2 * t + s2];
  }
}

// ---------------- launch ----------------
extern "C" void kernel_launch(void* const* d_in, const int* in_sizes, int n_in,
                              void* d_out, int out_size, void* d_ws, size_t ws_size,
                              hipStream_t stream) {
  (void)in_sizes; (void)n_in; (void)out_size; (void)ws_size;
  const float* hs     = (const float*)d_in[0];
  const float* ln1w   = (const float*)d_in[1];
  const float* qw     = (const float*)d_in[2];
  const float* kw     = (const float*)d_in[3];
  const float* vw     = (const float*)d_in[4];
  const float* ow     = (const float*)d_in[5];
  const float* reslnw = (const float*)d_in[6];
  const float* rw1    = (const float*)d_in[7];
  const float* rw3    = (const float*)d_in[8];
  const float* rw2    = (const float*)d_in[9];
  const float* postln = (const float*)d_in[10];
  const float* gatew  = (const float*)d_in[11];
  const float* ew1    = (const float*)d_in[12];
  const float* ew3    = (const float*)d_in[13];
  const float* ew2    = (const float*)d_in[14];
  float* out = (float*)d_out;

  char* ws = (char*)d_ws;
  auto alloc = [&](size_t bytes) -> char* {
    char* p = ws; ws += (bytes + 255) & ~(size_t)255; return p;
  };
  const int T = T_TOK;
  bhalf* hln1    = (bhalf*)alloc((size_t)T * H_DIM * 2);
  float* q32     = (float*)alloc((size_t)T * 1024 * 4);
  float* k32     = (float*)alloc((size_t)T * 256 * 4);
  float* v32     = (float*)alloc((size_t)T * 256 * 4);
  bhalf* qbf     = (bhalf*)alloc((size_t)T * 1024 * 2);
  bhalf* kbf     = (bhalf*)alloc((size_t)T * 256 * 2);
  bhalf* vbf     = (bhalf*)alloc((size_t)T * 256 * 2);
  bhalf* attnb   = (bhalf*)alloc((size_t)T * 1024 * 2);
  float* resattn = (float*)alloc((size_t)T * H_DIM * 4);
  bhalf* hres    = (bhalf*)alloc((size_t)T * H_DIM * 2);
  bhalf* hmidd   = (bhalf*)alloc((size_t)T * H_DIM * 2);
  bhalf* xpost   = (bhalf*)alloc((size_t)T * H_DIM * 2);
  bhalf* hmidm   = (bhalf*)alloc((size_t)(2 * T) * F_DIM * 2);
  int*   eid     = (int*)alloc((size_t)2 * T * 4);
  float* ewgt    = (float*)alloc((size_t)2 * T * 4);
  int*   cnt     = (int*)alloc(16 * 4);
  int*   off     = (int*)alloc(16 * 4);
  int*   cur     = (int*)alloc(16 * 4);
  int*   perm    = (int*)alloc((size_t)2 * T * 4);
  float* pwgt    = (float*)alloc((size_t)2 * T * 4);

  // 1) attention branch
  rmsnorm_kernel<<<T, 256, 0, stream>>>(hs, ln1w, hln1);
  gemm_kernel<<<dim3(1024 / BN, T / BM), 256, 0, stream>>>(hln1, H_DIM, qw, 1024, nullptr, q32, 1024, 1024, H_DIM);
  gemm_kernel<<<dim3(256 / BN, T / BM), 256, 0, stream>>>(hln1, H_DIM, kw, 256, nullptr, k32, 256, 256, H_DIM);
  gemm_kernel<<<dim3(256 / BN, T / BM), 256, 0, stream>>>(hln1, H_DIM, vw, 256, nullptr, v32, 256, 256, H_DIM);
  rope_pack_kernel<<<T, 256, 0, stream>>>(q32, qbf, NH_Q, 1);
  rope_pack_kernel<<<T, 256, 0, stream>>>(k32, kbf, NKV_H, 1);
  rope_pack_kernel<<<T, 256, 0, stream>>>(v32, vbf, NKV_H, 0);
  attn_kernel<<<dim3(S_LEN / 16, NH_Q, 2), 32, 0, stream>>>(qbf, kbf, vbf, attnb);
  gemm_kernel<<<dim3(H_DIM / BN, T / BM), 256, 0, stream>>>(attnb, 1024, ow, H_DIM, hs, resattn, H_DIM, H_DIM, 1024);

  // 2) dense parallel-residual SwiGLU MLP
  rmsnorm_kernel<<<T, 256, 0, stream>>>(resattn, reslnw, hres);
  dual_gemm_silu_kernel<<<dim3(H_DIM / BN, T / BM, 1), 256, 0, stream>>>(
      hres, H_DIM, nullptr, nullptr, nullptr, rw1, rw3, 0ULL, hmidd, H_DIM, T, H_DIM, H_DIM);
  gemm_kernel<<<dim3(H_DIM / BN, T / BM), 256, 0, stream>>>(hmidd, H_DIM, rw2, H_DIM, resattn, out, H_DIM, H_DIM, H_DIM);

  // 3) MoE branch on ORIGINAL layer input
  rmsnorm_kernel<<<T, 256, 0, stream>>>(hs, postln, xpost);
  zero16_kernel<<<1, 32, 0, stream>>>(cnt);
  gate_kernel<<<T, 32, 0, stream>>>(xpost, gatew, eid, ewgt, cnt);
  offsets_kernel<<<1, 32, 0, stream>>>(cnt, off, cur);
  scatter_kernel<<<(T + 255) / 256, 256, 0, stream>>>(eid, ewgt, off, cur, perm, pwgt);
  dual_gemm_silu_kernel<<<dim3(F_DIM / BN, T / BM, E_EXP), 256, 0, stream>>>(
      xpost, H_DIM, perm, off, cnt, ew1, ew3, (unsigned long long)H_DIM * F_DIM,
      hmidm, F_DIM, T, F_DIM, H_DIM);
  moe_gemm2_kernel<<<dim3(H_DIM / BN, T / BM, E_EXP), 256, 0, stream>>>(
      hmidm, F_DIM, perm, pwgt, off, cnt, ew2, (unsigned long long)F_DIM * H_DIM,
      out, H_DIM, H_DIM, F_DIM);
}
